// BandStructureModel_28217935135438
// MI455X (gfx1250) — matrix operand order repaired and checked by reading, not compiled, not run on hardware
//
#include <hip/hip_runtime.h>
#include <hip/hip_bf16.h>

// ---------------------------------------------------------------------------
// BandStructureModel on gfx1250 (MI455X): all 64-wide GEMMs through
// v_wmma_f32_16x16x32_f16 (f16 A/B, f32 accum), tiny K=1/2 layers + reductions
// in wave32 VALU. Weights staged transposed [fo][fi] as f16 in LDS so B
// fragments are 32-bit DS loads. One wave owns one 16-row WMMA M-tile.
// GEMM loops are kchunk-outer with multi-accumulator arrays so each A fragment
// is loaded from LDS exactly once per kchunk (DS bandwidth is the feed limit
// for XDL-class WMMA; independent accumulators also hide WMMA->WMMA hazards).
// ---------------------------------------------------------------------------

#define EMBEDD 64
#define NATOM  4
#define BATCH  8192
#define KPTS   128
#define NBAND  30
#define TWO_PI_F 6.283185307179586f

typedef _Float16 h16   __attribute__((ext_vector_type(16)));
typedef float    f32x8 __attribute__((ext_vector_type(8)));

union HFrag { h16 v; unsigned u[8]; };

// A fragment: 16x32 f16 tile, row-major in LDS with `stride` halves per row.
// Lane L: row = L%16; K pairs per VGPR per ISA 16-bit A layout.
__device__ __forceinline__ h16 lds_load_a(const _Float16* base, int stride) {
  const int lane = threadIdx.x & 31;
  const int row  = lane & 15;
  const int hi   = lane >> 4;
  HFrag f;
#pragma unroll
  for (int v = 0; v < 8; ++v) {
    const int k = ((v & 3) << 1) + ((v >> 2) << 4) + (hi << 3);
    f.u[v] = *reinterpret_cast<const unsigned*>(base + row * stride + k);
  }
  return f.v;
}

// B fragment: 32x16 (KxN) tile from weights stored transposed [N][K] in LDS.
// Lane L: col n = L%16; K pairs per VGPR per ISA 16-bit layout (same as A).
__device__ __forceinline__ h16 lds_load_b(const _Float16* wT, int stride) {
  const int lane = threadIdx.x & 31;
  const int n    = lane & 15;
  const int hi   = lane >> 4;
  HFrag f;
#pragma unroll
  for (int v = 0; v < 8; ++v) {
    const int k = ((v & 3) << 1) + ((v >> 2) << 4) + (hi << 3);
    f.u[v] = *reinterpret_cast<const unsigned*>(wT + n * stride + k);
  }
  return f.v;
}

__device__ __forceinline__ f32x8 wmma_f16(h16 a, h16 b, f32x8 c) {
  return __builtin_amdgcn_wmma_f32_16x16x32_f16(false, a, false, b,
                                                (short)0, c, false, false);
}

// C/D layout: lane L holds col n = L%16, VGPR v holds row m = v + 8*(L>=16).
__device__ __forceinline__ void lds_store_c_f16(_Float16* base, int stride, f32x8 c) {
  const int lane = threadIdx.x & 31;
  const int n    = lane & 15;
  const int hi   = lane >> 4;
#pragma unroll
  for (int v = 0; v < 8; ++v)
    base[(v + (hi << 3)) * stride + n] = (_Float16)c[v];
}

// Global f32 weights (fi,fo) row-major -> LDS f16 transposed [foPad][fi].
__device__ __forceinline__ void stage_wT(_Float16* wT, const float* w,
                                         int fi, int fo, int foPad) {
  __builtin_prefetch(w, 0, 1);
  for (int idx = threadIdx.x; idx < foPad * fi; idx += blockDim.x) {
    const int n = idx / fi, k = idx - n * fi;
    wT[idx] = (n < fo) ? (_Float16)w[k * fo + n] : (_Float16)0.f;
  }
}

__device__ __forceinline__ void stage_vec(float* dst, const float* src, int n, int npad) {
  for (int i = threadIdx.x; i < npad; i += blockDim.x)
    dst[i] = (i < n) ? src[i] : 0.f;
}

// ---------------------------------------------------------------------------
// Kernel 1: node features (embed + pos MLP), dist, pair mask.  VALU only
// (K=2 first layer; 0.1% of total FLOPs).  One 64-thread block per batch.
// ---------------------------------------------------------------------------
__global__ void __launch_bounds__(64)
k_init(const int* __restrict__ atom_types, const float* __restrict__ positions,
       const float* __restrict__ cell_length, const float* __restrict__ embed,
       const float* __restrict__ pw1, const float* __restrict__ pb1,
       const float* __restrict__ pw2, const float* __restrict__ pb2,
       float* __restrict__ nf, float* __restrict__ dist,
       float* __restrict__ pairm, float* __restrict__ maskw) {
  const int b = blockIdx.x;
  const int f = threadIdx.x;  // 0..63
  __shared__ float hsh[NATOM][EMBEDD];
  __shared__ float psh[NATOM];
  __shared__ float msh[NATOM];
  __shared__ int   tsh[NATOM];
  const float L = cell_length[b];
  if (f < NATOM) {
    const int t = atom_types[b * NATOM + f];
    tsh[f] = t;
    msh[f] = (t >= 0) ? 1.f : 0.f;
    psh[f] = positions[b * NATOM + f];
  }
  __syncthreads();
#pragma unroll
  for (int a = 0; a < NATOM; ++a) {
    const float pn = psh[a] / L;
    const float s = sinf(TWO_PI_F * pn), c = cosf(TWO_PI_F * pn);
    hsh[a][f] = fmaxf(s * pw1[f] + c * pw1[EMBEDD + f] + pb1[f], 0.f);
  }
  __syncthreads();
#pragma unroll
  for (int a = 0; a < NATOM; ++a) {
    float acc = pb2[f];
    for (int k = 0; k < EMBEDD; ++k) acc += hsh[a][k] * pw2[k * EMBEDD + f];
    const int t = tsh[a];
    const float e = (t >= 0) ? embed[t * EMBEDD + f] : 0.f;
    nf[((size_t)b * NATOM + a) * EMBEDD + f] = e + acc;
  }
  if (f < 16) {
    const int i = f >> 2, j = f & 3;
    const float d = fabsf(psh[i] - psh[j]);
    dist[b * 16 + f]  = fminf(d, L - d) / L;
    pairm[b * 16 + f] = (i != j) ? msh[i] * msh[j] : 0.f;
  }
  if (f < NATOM) maskw[b * NATOM + f] = msh[f];
}

// ---------------------------------------------------------------------------
// Kernel 2: one GNN layer.  One wave per batch element; the 16 (i,j) pairs
// form exactly one WMMA M-tile.  4 waves (128 thr) per workgroup.
// ---------------------------------------------------------------------------
__global__ void __launch_bounds__(128)
k_layer(const float* ew1, const float* eb1, const float* ew2, const float* eb2,
        const float* mw1, const float* mb1, const float* mw2, const float* mb2,
        const float* uw1, const float* ub1, const float* uw2, const float* ub2,
        const float* lngp, const float* lnbp,
        float* __restrict__ nf, const float* __restrict__ dist,
        const float* __restrict__ pairm, const float* __restrict__ maskw) {
  __shared__ _Float16 we2T[64 * 64];
  __shared__ _Float16 wm1T[64 * 192];
  __shared__ _Float16 wm2T[64 * 64];
  __shared__ _Float16 wu1T[64 * 128];
  __shared__ _Float16 wu2T[64 * 64];
  __shared__ float we1s[64], be1s[64], be2s[64], bm1s[64], bm2s[64];
  __shared__ float bu1s[64], bu2s[64], lngs[64], lnbs[64];
  __shared__ _Float16 astg[4][16 * 192];
  __shared__ _Float16 hstg[4][16 * 64];
  __shared__ float    fstg[4][256];

  stage_wT(we2T, ew2, 64, 64, 64);
  stage_wT(wm1T, mw1, 192, 64, 64);
  stage_wT(wm2T, mw2, 64, 64, 64);
  stage_wT(wu1T, uw1, 128, 64, 64);
  stage_wT(wu2T, uw2, 64, 64, 64);
  stage_vec(we1s, ew1, 64, 64);  stage_vec(be1s, eb1, 64, 64);
  stage_vec(be2s, eb2, 64, 64);
  stage_vec(bm1s, mb1, 64, 64);  stage_vec(bm2s, mb2, 64, 64);
  stage_vec(bu1s, ub1, 64, 64);  stage_vec(bu2s, ub2, 64, 64);
  stage_vec(lngs, lngp, 64, 64); stage_vec(lnbs, lnbp, 64, 64);
  __syncthreads();

  const int wave = threadIdx.x >> 5;
  const int lane = threadIdx.x & 31;
  const int hi   = lane >> 4;
  const int b    = blockIdx.x * 4 + wave;
  _Float16* A = astg[wave];
  _Float16* H = hstg[wave];
  float*    Fr = fstg[wave];

  // nf (4x64) into per-wave f32 scratch
  for (int i = lane; i < 256; i += 32) Fr[i] = nf[(size_t)b * 256 + i];

  // edge MLP layer1 (K=1) in VALU -> H[16x64]
  for (int i = lane; i < 16 * 64; i += 32) {
    const int p = i >> 6, f = i & 63;
    const float dd = dist[b * 16 + p];
    H[p * 64 + f] = (_Float16)fmaxf(dd * we1s[f] + be1s[f], 0.f);
  }
  // msg input cols 0..127: [nf[i] | nf[j]] for pair p = i*4+j
  for (int i = lane; i < 16 * 128; i += 32) {
    const int p = i >> 7, col = i & 127;
    const int atom = (col < 64) ? (p >> 2) : (p & 3);
    A[p * 192 + col] = (_Float16)Fr[atom * 64 + (col & 63)];
  }
  // edge MLP layer2 (64->64) WMMA -> msg input cols 128..191
  {
    f32x8 acc[4] = {};
#pragma unroll
    for (int kc = 0; kc < 2; ++kc) {
      const h16 a = lds_load_a(H + kc * 32, 64);
#pragma unroll
      for (int nt = 0; nt < 4; ++nt)
        acc[nt] = wmma_f16(a, lds_load_b(we2T + (nt * 16) * 64 + kc * 32, 64), acc[nt]);
    }
#pragma unroll
    for (int nt = 0; nt < 4; ++nt) {
      const float bb = be2s[nt * 16 + (lane & 15)];
      f32x8 c = acc[nt];
#pragma unroll
      for (int v = 0; v < 8; ++v) c[v] += bb;
      lds_store_c_f16(A + 128 + nt * 16, 192, c);
    }
  }
  // msg layer1 (192->64) WMMA, relu -> H
  {
    f32x8 acc[4] = {};
#pragma unroll
    for (int kc = 0; kc < 6; ++kc) {
      const h16 a = lds_load_a(A + kc * 32, 192);
#pragma unroll
      for (int nt = 0; nt < 4; ++nt)
        acc[nt] = wmma_f16(a, lds_load_b(wm1T + (nt * 16) * 192 + kc * 32, 192), acc[nt]);
    }
#pragma unroll
    for (int nt = 0; nt < 4; ++nt) {
      const float bb = bm1s[nt * 16 + (lane & 15)];
      f32x8 c = acc[nt];
#pragma unroll
      for (int v = 0; v < 8; ++v) c[v] = fmaxf(c[v] + bb, 0.f);
      lds_store_c_f16(H + nt * 16, 64, c);
    }
  }
  // msg layer2 (64->64) WMMA + pair mask + aggregate over j -> upd cols 64..127
  {
    float pr[8];
#pragma unroll
    for (int v = 0; v < 8; ++v) pr[v] = pairm[b * 16 + v + (hi << 3)];
    f32x8 acc[4] = {};
#pragma unroll
    for (int kc = 0; kc < 2; ++kc) {
      const h16 a = lds_load_a(H + kc * 32, 64);
#pragma unroll
      for (int nt = 0; nt < 4; ++nt)
        acc[nt] = wmma_f16(a, lds_load_b(wm2T + (nt * 16) * 64 + kc * 32, 64), acc[nt]);
    }
#pragma unroll
    for (int nt = 0; nt < 4; ++nt) {
      const float bb = bm2s[nt * 16 + (lane & 15)];
      float a0 = 0.f, a1 = 0.f;
#pragma unroll
      for (int v = 0; v < 4; ++v) a0 += (acc[nt][v] + bb) * pr[v];
#pragma unroll
      for (int v = 4; v < 8; ++v) a1 += (acc[nt][v] + bb) * pr[v];
      const int col = 64 + nt * 16 + (lane & 15);
      A[(2 * hi + 0) * 192 + col] = (_Float16)a0;  // agg rows = atoms 0..3
      A[(2 * hi + 1) * 192 + col] = (_Float16)a1;
    }
  }
  // upd input cols 0..63 rows 0..3 = nf
  for (int i = lane; i < 4 * 64; i += 32)
    A[(i >> 6) * 192 + (i & 63)] = (_Float16)Fr[i];
  // upd layer1 (128->64) WMMA, relu -> H
  {
    f32x8 acc[4] = {};
#pragma unroll
    for (int kc = 0; kc < 4; ++kc) {
      const h16 a = lds_load_a(A + kc * 32, 192);
#pragma unroll
      for (int nt = 0; nt < 4; ++nt)
        acc[nt] = wmma_f16(a, lds_load_b(wu1T + (nt * 16) * 128 + kc * 32, 128), acc[nt]);
    }
#pragma unroll
    for (int nt = 0; nt < 4; ++nt) {
      const float bb = bu1s[nt * 16 + (lane & 15)];
      f32x8 c = acc[nt];
#pragma unroll
      for (int v = 0; v < 8; ++v) c[v] = fmaxf(c[v] + bb, 0.f);
      lds_store_c_f16(H + nt * 16, 64, c);
    }
  }
  // upd layer2 (64->64) WMMA; accumulate nf+upd for atoms 0..3 (rows 0..3)
  {
    f32x8 acc[4] = {};
#pragma unroll
    for (int kc = 0; kc < 2; ++kc) {
      const h16 a = lds_load_a(H + kc * 32, 64);
#pragma unroll
      for (int nt = 0; nt < 4; ++nt)
        acc[nt] = wmma_f16(a, lds_load_b(wu2T + (nt * 16) * 64 + kc * 32, 64), acc[nt]);
    }
    if (hi == 0) {
#pragma unroll
      for (int nt = 0; nt < 4; ++nt) {
        const float bb = bu2s[nt * 16 + (lane & 15)];
#pragma unroll
        for (int v = 0; v < 4; ++v)
          Fr[v * 64 + nt * 16 + (lane & 15)] += acc[nt][v] + bb;
      }
    }
  }
  // layernorm per atom with wave32 shuffle reduction; write back nf
  for (int a = 0; a < NATOM; ++a) {
    const float x0 = Fr[a * 64 + lane];
    const float x1 = Fr[a * 64 + 32 + lane];
    float s = x0 + x1, ss = x0 * x0 + x1 * x1;
#pragma unroll
    for (int off = 16; off >= 1; off >>= 1) {
      s  += __shfl_xor(s, off, 32);
      ss += __shfl_xor(ss, off, 32);
    }
    const float m   = s * (1.f / 64.f);
    const float inv = rsqrtf(ss * (1.f / 64.f) - m * m + 1e-5f);
    const float mv  = maskw[b * NATOM + a];
    nf[(size_t)b * 256 + a * 64 + lane] =
        ((x0 - m) * inv * lngs[lane] + lnbs[lane]) * mv;
    nf[(size_t)b * 256 + a * 64 + 32 + lane] =
        ((x1 - m) * inv * lngs[32 + lane] + lnbs[32 + lane]) * mv;
  }
}

// ---------------------------------------------------------------------------
// Kernel 3: attention pooling + cell MLP -> xgp (B,256).  VALU (tiny).
// ---------------------------------------------------------------------------
__global__ void __launch_bounds__(64)
k_pool(const float* __restrict__ nf, const float* __restrict__ maskw,
       const float* __restrict__ cell_length,
       const float* aw1, const float* ab1, const float* aw2, const float* ab2,
       const float* cw1, const float* cb1, const float* cw2, const float* cb2,
       float* __restrict__ xgp) {
  const int b = blockIdx.x;
  const int f = threadIdx.x;  // 0..63
  __shared__ float nsh[NATOM][EMBEDD];
  __shared__ float hsh[NATOM][32];
  __shared__ float ssh[NATOM];
  __shared__ float msh[NATOM];
  __shared__ float csh[EMBEDD];
  for (int i = f; i < 256; i += 64) nsh[i >> 6][i & 63] = nf[(size_t)b * 256 + i];
  if (f < NATOM) msh[f] = maskw[b * NATOM + f];
  __syncthreads();
  if (f < 32) {
#pragma unroll
    for (int a = 0; a < NATOM; ++a) {
      float acc = ab1[f];
      for (int k = 0; k < EMBEDD; ++k) acc += nsh[a][k] * aw1[k * 32 + f];
      hsh[a][f] = fmaxf(acc, 0.f);
    }
  }
  __syncthreads();
  if (f < NATOM) {
    float acc = ab2[0];
    for (int j = 0; j < 32; ++j) acc += hsh[f][j] * aw2[j];
    ssh[f] = (msh[f] == 0.f) ? -1e9f : acc;
  }
  __syncthreads();
  const float mxs = fmaxf(fmaxf(ssh[0], ssh[1]), fmaxf(ssh[2], ssh[3]));
  float e[NATOM], Z = 0.f;
#pragma unroll
  for (int a = 0; a < NATOM; ++a) { e[a] = expf(ssh[a] - mxs); Z += e[a]; }
  float wsum = 0.f, vmax = -1e9f, sm = 0.f, cnt = 0.f;
#pragma unroll
  for (int a = 0; a < NATOM; ++a) {
    const float v = nsh[a][f];
    wsum += v * e[a] / Z;
    vmax = fmaxf(vmax, (msh[a] == 0.f) ? -1e9f : v);
    sm += v;
    cnt += msh[a];
  }
  cnt = fmaxf(cnt, 1.f);
  xgp[(size_t)b * 256 + f]        = wsum;
  xgp[(size_t)b * 256 + 64 + f]   = vmax;
  xgp[(size_t)b * 256 + 128 + f]  = sm / cnt;
  // cell MLP (1->64->64)
  const float L = cell_length[b];
  csh[f] = fmaxf(L * cw1[f] + cb1[f], 0.f);
  __syncthreads();
  float acc = cb2[f];
  for (int k = 0; k < EMBEDD; ++k) acc += csh[k] * cw2[k * EMBEDD + f];
  xgp[(size_t)b * 256 + 192 + f] = acc;
}

// ---------------------------------------------------------------------------
// Kernel 4: global-property MLP as a real GEMM: (B,256)->128 relu ->128.
// 2 waves per block, 16 rows per wave; kc-outer with 8 live accumulators.
// ---------------------------------------------------------------------------
__global__ void __launch_bounds__(64)
k_gp(const float* __restrict__ xgp, const float* gw1, const float* gb1,
     const float* gw2, const float* gb2, float* __restrict__ pg) {
  __shared__ _Float16 wg1T[128 * 256];
  __shared__ _Float16 wg2T[128 * 128];
  __shared__ float b1s[128], b2s[128];
  __shared__ _Float16 astg[2][16 * 256];
  __shared__ _Float16 hstg[2][16 * 128];

  stage_wT(wg1T, gw1, 256, 128, 128);
  stage_wT(wg2T, gw2, 128, 128, 128);
  stage_vec(b1s, gb1, 128, 128);
  stage_vec(b2s, gb2, 128, 128);
  __syncthreads();

  const int wave = threadIdx.x >> 5;
  const int lane = threadIdx.x & 31;
  const int hi   = lane >> 4;
  const int row0 = blockIdx.x * 32 + wave * 16;
  _Float16* A = astg[wave];
  _Float16* H = hstg[wave];

  for (int i = lane; i < 16 * 256; i += 32)
    A[i] = (_Float16)xgp[(size_t)row0 * 256 + i];

  {
    f32x8 acc[8] = {};
#pragma unroll
    for (int kc = 0; kc < 8; ++kc) {
      const h16 a = lds_load_a(A + kc * 32, 256);
#pragma unroll
      for (int nt = 0; nt < 8; ++nt)
        acc[nt] = wmma_f16(a, lds_load_b(wg1T + (nt * 16) * 256 + kc * 32, 256), acc[nt]);
    }
#pragma unroll
    for (int nt = 0; nt < 8; ++nt) {
      const float bb = b1s[nt * 16 + (lane & 15)];
      f32x8 c = acc[nt];
#pragma unroll
      for (int v = 0; v < 8; ++v) c[v] = fmaxf(c[v] + bb, 0.f);
      lds_store_c_f16(H + nt * 16, 128, c);
    }
  }
  {
    f32x8 acc[8] = {};
#pragma unroll
    for (int kc = 0; kc < 4; ++kc) {
      const h16 a = lds_load_a(H + kc * 32, 128);
#pragma unroll
      for (int nt = 0; nt < 8; ++nt)
        acc[nt] = wmma_f16(a, lds_load_b(wg2T + (nt * 16) * 128 + kc * 32, 128), acc[nt]);
    }
#pragma unroll
    for (int nt = 0; nt < 8; ++nt) {
      const float bb = b2s[nt * 16 + (lane & 15)];
#pragma unroll
      for (int v = 0; v < 8; ++v)
        pg[(size_t)(row0 + v + (hi << 3)) * 128 + nt * 16 + (lane & 15)] = acc[nt][v] + bb;
    }
  }
}

// ---------------------------------------------------------------------------
// Kernel 5: band head. One block = (batch, 64 k-points); 4 waves x 16 rows.
// pg columns are identical for every row and every wave -> one block-shared
// 16x128 pgA tile; per-wave only the 16x64 kf tile.  kin layer1 (K=2) in
// VALU; kf 64->64, band 192->128->64->32(30) via WMMA, kc-outer multi-acc.
// ---------------------------------------------------------------------------
__global__ void __launch_bounds__(128)
k_band(const float* __restrict__ k_points, const float* __restrict__ pg,
       const float* kw1, const float* kb1, const float* kw2, const float* kb2,
       const float* bw1, const float* bb1g, const float* bw2, const float* bb2g,
       const float* bw3, const float* bb3g, float* __restrict__ out) {
  __shared__ _Float16 wk2T[64 * 64];
  __shared__ _Float16 wb1T[128 * 192];
  __shared__ _Float16 wb2T[64 * 128];
  __shared__ _Float16 wb3T[32 * 64];
  __shared__ float kw1s[128], kb1s[64], bk2[64], bb1[128], bb2[64], bb3[32];
  __shared__ _Float16 pgA[16 * 128];       // block-shared A cols 0..127
  __shared__ _Float16 kfstg[4][16 * 64];   // per-wave A cols 128..191 / l3 in
  __shared__ _Float16 hstg[4][16 * 128];

  const int b     = blockIdx.x >> 1;
  const int khalf = (blockIdx.x & 1) << 6;

  stage_wT(wk2T, kw2, 64, 64, 64);
  stage_wT(wb1T, bw1, 192, 128, 128);
  stage_wT(wb2T, bw2, 128, 64, 64);
  stage_wT(wb3T, bw3, 64, NBAND, 32);
  stage_vec(kw1s, kw1, 128, 128);
  stage_vec(kb1s, kb1, 64, 64);
  stage_vec(bk2, kb2, 64, 64);
  stage_vec(bb1, bb1g, 128, 128);
  stage_vec(bb2, bb2g, 64, 64);
  stage_vec(bb3, bb3g, NBAND, 32);
  for (int i = threadIdx.x; i < 16 * 128; i += blockDim.x)
    pgA[i] = (_Float16)pg[(size_t)b * 128 + (i & 127)];
  __syncthreads();

  const int wave = threadIdx.x >> 5;
  const int lane = threadIdx.x & 31;
  const int hi   = lane >> 4;
  const int kb0  = khalf + wave * 16;
  _Float16* KF = kfstg[wave];
  _Float16* H  = hstg[wave];

  // k MLP layer1 (K=2: sin/cos) in VALU -> H[16x64] (stride 128)
  for (int i = lane; i < 16 * 64; i += 32) {
    const int r = i >> 6, f = i & 63;
    const float kp = k_points[(size_t)b * KPTS + kb0 + r];
    const float s = sinf(TWO_PI_F * kp), c = cosf(TWO_PI_F * kp);
    H[r * 128 + f] = (_Float16)fmaxf(s * kw1s[f] + c * kw1s[64 + f] + kb1s[f], 0.f);
  }
  // k MLP layer2 (64->64) WMMA -> KF (x cols 128..191)
  {
    f32x8 acc[4] = {};
#pragma unroll
    for (int kc = 0; kc < 2; ++kc) {
      const h16 a = lds_load_a(H + kc * 32, 128);
#pragma unroll
      for (int nt = 0; nt < 4; ++nt)
        acc[nt] = wmma_f16(a, lds_load_b(wk2T + (nt * 16) * 64 + kc * 32, 64), acc[nt]);
    }
#pragma unroll
    for (int nt = 0; nt < 4; ++nt) {
      const float bb = bk2[nt * 16 + (lane & 15)];
      f32x8 c = acc[nt];
#pragma unroll
      for (int v = 0; v < 8; ++v) c[v] += bb;
      lds_store_c_f16(KF + nt * 16, 64, c);
    }
  }
  // band layer1 (192->128): A = [pgA | KF], 8 live accumulators -> H
  {
    f32x8 acc[8] = {};
#pragma unroll
    for (int kc = 0; kc < 6; ++kc) {
      const h16 a = (kc < 4) ? lds_load_a(pgA + kc * 32, 128)
                             : lds_load_a(KF + (kc - 4) * 32, 64);
#pragma unroll
      for (int nt = 0; nt < 8; ++nt)
        acc[nt] = wmma_f16(a, lds_load_b(wb1T + (nt * 16) * 192 + kc * 32, 192), acc[nt]);
    }
#pragma unroll
    for (int nt = 0; nt < 8; ++nt) {
      const float bb = bb1[nt * 16 + (lane & 15)];
      f32x8 c = acc[nt];
#pragma unroll
      for (int v = 0; v < 8; ++v) c[v] = fmaxf(c[v] + bb, 0.f);
      lds_store_c_f16(H + nt * 16, 128, c);
    }
  }
  // band layer2 (128->64) WMMA, relu -> KF (kf consumed by layer1)
  {
    f32x8 acc[4] = {};
#pragma unroll
    for (int kc = 0; kc < 4; ++kc) {
      const h16 a = lds_load_a(H + kc * 32, 128);
#pragma unroll
      for (int nt = 0; nt < 4; ++nt)
        acc[nt] = wmma_f16(a, lds_load_b(wb2T + (nt * 16) * 128 + kc * 32, 128), acc[nt]);
    }
#pragma unroll
    for (int nt = 0; nt < 4; ++nt) {
      const float bb = bb2[nt * 16 + (lane & 15)];
      f32x8 c = acc[nt];
#pragma unroll
      for (int v = 0; v < 8; ++v) c[v] = fmaxf(c[v] + bb, 0.f);
      lds_store_c_f16(KF + nt * 16, 64, c);
    }
  }
  // band layer3 (64->32, keep 30) WMMA -> global out
  {
    f32x8 acc[2] = {};
#pragma unroll
    for (int kc = 0; kc < 2; ++kc) {
      const h16 a = lds_load_a(KF + kc * 32, 64);
#pragma unroll
      for (int nt = 0; nt < 2; ++nt)
        acc[nt] = wmma_f16(a, lds_load_b(wb3T + (nt * 16) * 64 + kc * 32, 64), acc[nt]);
    }
#pragma unroll
    for (int nt = 0; nt < 2; ++nt) {
      const int n = nt * 16 + (lane & 15);
      if (n < NBAND) {
        const float bb = bb3[n];
#pragma unroll
        for (int v = 0; v < 8; ++v) {
          const int kk = kb0 + v + (hi << 3);
          out[((size_t)b * KPTS + kk) * NBAND + n] = acc[nt][v] + bb;
        }
      }
    }
  }
}

// ---------------------------------------------------------------------------
// Launch.  Input order: flat depth-first insertion order of setup_inputs():
//   0 atom_types, 1 positions, 2 cell_length, 3 k_points, then params:
//   4 embed; 5-8 pos(w1,b1,w2,b2);
//   9+14*l: layer l -> edge(w1,b1,w2,b2), msg(4), upd(4), ln_g, ln_b;
//   51-54 attn; 55-58 cell; 59-62 k; 63-66 gp; 67-72 band(w1,b1,w2,b2,w3,b3).
// ---------------------------------------------------------------------------
extern "C" void kernel_launch(void* const* d_in, const int* in_sizes, int n_in,
                              void* d_out, int out_size, void* d_ws, size_t ws_size,
                              hipStream_t stream) {
  (void)in_sizes; (void)out_size; (void)ws_size;
  const int* atom_types      = (const int*)d_in[0];
  const float* positions     = (const float*)d_in[1];
  const float* cell_length   = (const float*)d_in[2];
  const float* k_points      = (const float*)d_in[3];
  const float* P[80];
  for (int i = 4; i < n_in && i < 80; ++i) P[i] = (const float*)d_in[i];

  float* ws    = (float*)d_ws;
  float* nf    = ws;                             // B*256
  float* dist  = nf    + (size_t)BATCH * 256;    // B*16
  float* pairm = dist  + (size_t)BATCH * 16;     // B*16
  float* maskw = pairm + (size_t)BATCH * 16;     // B*4
  float* xgp   = maskw + (size_t)BATCH * 4;      // B*256
  float* pg    = xgp   + (size_t)BATCH * 256;    // B*128
  float* out   = (float*)d_out;

  k_init<<<BATCH, 64, 0, stream>>>(atom_types, positions, cell_length,
      P[4], P[5], P[6], P[7], P[8], nf, dist, pairm, maskw);

  for (int l = 0; l < 3; ++l) {
    const int base = 9 + l * 14;
    k_layer<<<BATCH / 4, 128, 0, stream>>>(
        P[base + 0], P[base + 1], P[base + 2], P[base + 3],
        P[base + 4], P[base + 5], P[base + 6], P[base + 7],
        P[base + 8], P[base + 9], P[base + 10], P[base + 11],
        P[base + 12], P[base + 13],
        nf, dist, pairm, maskw);
  }

  k_pool<<<BATCH, 64, 0, stream>>>(nf, maskw, cell_length,
      P[51], P[52], P[53], P[54], P[55], P[56], P[57], P[58], xgp);

  k_gp<<<BATCH / 32, 64, 0, stream>>>(xgp, P[63], P[64], P[65], P[66], pg);

  k_band<<<BATCH * 2, 128, 0, stream>>>(k_points, pg,
      P[59], P[60], P[61], P[62],
      P[67], P[68], P[69], P[70], P[71], P[72], out);
}
